// gcgru_91190745628777
// MI455X (gfx1250) — matrix operand order repaired and checked by reading, not compiled, not run on hardware
//
#include <hip/hip_runtime.h>
#include <cmath>

typedef __bf16 bf16;
typedef __attribute__((ext_vector_type(8)))  __bf16 v8bf;
typedef __attribute__((ext_vector_type(16))) __bf16 v16bf;
typedef __attribute__((ext_vector_type(8)))  float  v8f;
typedef __attribute__((ext_vector_type(4)))  int    v4i;

#define BATCH  512
#define NNODES 4096
#define UNITS  256

// ---------------------------------------------------------------------------
// Async global->LDS copy (CDNA5 GLOBAL_LOAD_ASYNC_TO_LDS_B128, ASYNCcnt path)
// with a synchronous fallback if the builtin is absent on this toolchain.
// Builtin signature (from hipcc diagnostic): (global v4i*, local v4i*, Ii, Ii)
// ---------------------------------------------------------------------------
#if __has_builtin(__builtin_amdgcn_global_load_async_to_lds_b128)
#define ASYNC_COPY 1
#else
#define ASYNC_COPY 0
#endif

typedef __attribute__((address_space(1))) v4i gas_v4i;   // global
typedef __attribute__((address_space(3))) v4i las_v4i;   // LDS

__device__ __forceinline__ void async_copy16(const bf16* g, bf16* l)
{
#if ASYNC_COPY
    v4i* gp = (v4i*)const_cast<bf16*>(g);
    v4i* lp = (v4i*)l;
    __builtin_amdgcn_global_load_async_to_lds_b128(
        (gas_v4i*)gp, (las_v4i*)lp, 0, 0);
#else
    *(v8bf*)l = *(const v8bf*)g;
#endif
}

template <int N>
__device__ __forceinline__ void async_wait()
{
#if ASYNC_COPY
#if __has_builtin(__builtin_amdgcn_s_wait_asynccnt)
    __builtin_amdgcn_s_wait_asynccnt(N);
#else
    asm volatile("s_wait_asynccnt %0" :: "n"(N) : "memory");
#endif
#endif
}

// ---------------------------------------------------------------------------
// 1) adj_mix = wa0*adj0 + wa1*adj1 + wa2*adj2, fused fp32 -> bf16 (201MB->32MB)
// ---------------------------------------------------------------------------
__global__ __launch_bounds__(256) void k_adj_mix(const float* __restrict__ adj,
                                                 const float* __restrict__ wa,
                                                 bf16* __restrict__ out)
{
    const int nn = NNODES * NNODES;
    int i = blockIdx.x * 256 + threadIdx.x;          // float4 index
    float w0 = wa[0], w1 = wa[1], w2 = wa[2];
    float4 a0 = ((const float4*)adj)[i];
    float4 a1 = ((const float4*)(adj + nn))[i];
    float4 a2 = ((const float4*)(adj + 2 * nn))[i];
    int b = i * 4;
    out[b + 0] = (bf16)(w0 * a0.x + w1 * a1.x + w2 * a2.x);
    out[b + 1] = (bf16)(w0 * a0.y + w1 * a1.y + w2 * a2.y);
    out[b + 2] = (bf16)(w0 * a0.z + w1 * a1.z + w2 * a2.z);
    out[b + 3] = (bf16)(w0 * a0.w + w1 * a1.w + w2 * a2.w);
}

// ---------------------------------------------------------------------------
// 2) generic fp32 -> bf16 convert (n multiple of 1024, grid = n/1024)
// ---------------------------------------------------------------------------
__global__ __launch_bounds__(256) void k_cvt_bf16(const float* __restrict__ s,
                                                  bf16* __restrict__ d)
{
    int i = blockIdx.x * 256 + threadIdx.x;
    float4 v = ((const float4*)s)[i];
    int b = i * 4;
    d[b + 0] = (bf16)v.x;
    d[b + 1] = (bf16)v.y;
    d[b + 2] = (bf16)v.z;
    d[b + 3] = (bf16)v.w;
}

// ---------------------------------------------------------------------------
// 3) bf16 GEMM: C[M,N] = (relu?) A[M,K] * B[K,N], f32 accumulate, bf16 out.
//    Block tile 128x128, BK=32 double-buffered LDS via async-to-LDS copies,
//    8 waves (4x2) of 32x64 -> 8 back-to-back WMMAs per wave per K-step.
// ---------------------------------------------------------------------------
__device__ __forceinline__ void load_tiles_async(const bf16* __restrict__ A,
                                                 const bf16* __restrict__ B,
                                                 int K, int N, int bm, int bn,
                                                 int kt, int tid,
                                                 bf16 (*lA)[40], bf16 (*lB)[136])
{
    #pragma unroll
    for (int j = 0; j < 2; ++j) {
        int i = tid + j * 256;
        int r = i >> 2, c = (i & 3) * 8;
        async_copy16(&A[(size_t)(bm + r) * K + kt * 32 + c], &lA[r][c]);
    }
    #pragma unroll
    for (int j = 0; j < 2; ++j) {
        int i = tid + j * 256;
        int r = i >> 4, c = (i & 15) * 8;
        async_copy16(&B[(size_t)(kt * 32 + r) * N + bn + c], &lB[r][c]);
    }
}

template <bool RELU>
__global__ __launch_bounds__(256) void gemm_bf16(const bf16* __restrict__ A,
                                                 const bf16* __restrict__ B,
                                                 bf16* __restrict__ C,
                                                 int M, int N, int K)
{
    __shared__ bf16 lA[2][128][40];   // 32 cols + 8 pad (row stride 80B, 16B aligned)
    __shared__ bf16 lB[2][32][136];   // 128 cols + 8 pad (row stride 272B, 16B aligned)

    const int tid  = threadIdx.x;
    const int lane = tid & 31;
    const int wave = tid >> 5;
    const int wm   = wave >> 1;       // 0..3 : wave row (32 rows each)
    const int wn   = wave & 1;        // 0..1 : wave col (64 cols each)
    const int lh   = lane >> 4;       // lane half
    const int l15  = lane & 15;

    const int bm = blockIdx.y * 128;
    const int bn = blockIdx.x * 128;

    v8f acc[2][4];
    #pragma unroll
    for (int m = 0; m < 2; ++m)
        #pragma unroll
        for (int n = 0; n < 4; ++n)
            #pragma unroll
            for (int e = 0; e < 8; ++e) acc[m][n][e] = 0.f;

    const int nk = K >> 5;

    // prologue: async-load tile 0 into buffer 0
    load_tiles_async(A, B, K, N, bm, bn, 0, tid, lA[0], lB[0]);

    for (int kt = 0; kt < nk; ++kt) {
        const int  buf = kt & 1;
        const bool pre = (kt + 1) < nk;

        if (pre) {
            // async prefetch tile kt+1 into the other buffer (ASYNCcnt += 4)
            load_tiles_async(A, B, K, N, bm, bn, kt + 1, tid,
                             lA[buf ^ 1], lB[buf ^ 1]);
            async_wait<4>();          // tile kt's 4 copies are complete (in-order)
        } else {
            async_wait<0>();
        }
        __syncthreads();              // everyone's tile-kt copies are visible

        // ---- load all fragments first, then 8 back-to-back WMMAs ----
        // A fragment, ISA 16-bit 16x32 layout:
        //   lanes 0-15 : M=l15, K 0-7 / 16-23 ; lanes 16-31: K 8-15 / 24-31
        v16bf af[2], bfr[4];
        #pragma unroll
        for (int m = 0; m < 2; ++m) {
            const bf16* p = &lA[buf][wm * 32 + m * 16 + l15][lh * 8];
            ((v8bf*)&af[m])[0] = *(const v8bf*)p;
            ((v8bf*)&af[m])[1] = *(const v8bf*)(p + 16);
        }
        #pragma unroll
        for (int n = 0; n < 4; ++n) {
            // B fragment 32x16: N=l15, lanes 0-15 K 0-7 / 16-23 ; lanes 16-31 K 8-15 / 24-31
            const bf16* p = &lB[buf][lh * 8][wn * 64 + n * 16 + l15];
            #pragma unroll
            for (int s = 0; s < 16; ++s) bfr[n][s] = p[(s < 8 ? s : s + 8) * 136];
        }
        #pragma unroll
        for (int n = 0; n < 4; ++n)
            #pragma unroll
            for (int m = 0; m < 2; ++m)
                acc[m][n] = __builtin_amdgcn_wmma_f32_16x16x32_bf16(
                    false, af[m], false, bfr[n], (short)0, acc[m][n], false, false);

        __syncthreads();              // all waves done with buf before it is refilled
    }

    // epilogue: C/D layout — VGPR r: lanes 0-15 M=r, lanes 16-31 M=r+8, N=l15
    #pragma unroll
    for (int m = 0; m < 2; ++m)
        #pragma unroll
        for (int n = 0; n < 4; ++n)
            #pragma unroll
            for (int r = 0; r < 8; ++r) {
                float v = acc[m][n][r];
                if (RELU) v = v > 0.f ? v : 0.f;
                int row = bm + wm * 32 + m * 16 + lh * 8 + r;
                int col = bn + wn * 64 + n * 16 + l15;
                C[(size_t)row * N + col] = (bf16)v;
            }
}

// ---------------------------------------------------------------------------
// 4) fused GRU gates: z = sig(x@Wz + s@Uz + bz), h = tanh(x@Wh + s@Uh + bh)
//    out = (1-z)*state + z*h.   M=512, N=K=256. Same WMMA tiling, dual accums.
// ---------------------------------------------------------------------------
__global__ __launch_bounds__(256) void gru_kernel(const bf16* __restrict__ Xb,
                                                  const bf16* __restrict__ Sb,
                                                  const bf16* __restrict__ Wz,
                                                  const bf16* __restrict__ Uz,
                                                  const bf16* __restrict__ Wh,
                                                  const bf16* __restrict__ Uh,
                                                  const float* __restrict__ bz,
                                                  const float* __restrict__ bh,
                                                  const float* __restrict__ state,
                                                  float* __restrict__ out)
{
    __shared__ bf16 lX[128][40];
    __shared__ bf16 lS[128][40];
    __shared__ bf16 lW[4][32][136];   // Wz, Uz, Wh, Uh tiles

    const int tid  = threadIdx.x;
    const int lane = tid & 31;
    const int wave = tid >> 5;
    const int wm   = wave >> 1;
    const int wn   = wave & 1;
    const int lh   = lane >> 4;
    const int l15  = lane & 15;
    const int bm   = blockIdx.y * 128;
    const int bn   = blockIdx.x * 128;
    const int N = UNITS, K = UNITS;

    v8f accz[2][4], acch[2][4];
    #pragma unroll
    for (int m = 0; m < 2; ++m)
        #pragma unroll
        for (int n = 0; n < 4; ++n)
            #pragma unroll
            for (int e = 0; e < 8; ++e) { accz[m][n][e] = 0.f; acch[m][n][e] = 0.f; }

    const bf16* Ws[4] = {Wz, Uz, Wh, Uh};

    for (int kt = 0; kt < (K >> 5); ++kt) {
        // all tile loads issued asynchronously, one wait, one barrier
        #pragma unroll
        for (int j = 0; j < 2; ++j) {
            int i = tid + j * 256;
            int r = i >> 2, c = (i & 3) * 8;
            async_copy16(&Xb[(size_t)(bm + r) * K + kt * 32 + c], &lX[r][c]);
            async_copy16(&Sb[(size_t)(bm + r) * K + kt * 32 + c], &lS[r][c]);
        }
        #pragma unroll
        for (int w = 0; w < 4; ++w)
            #pragma unroll
            for (int j = 0; j < 2; ++j) {
                int i = tid + j * 256;
                int r = i >> 4, c = (i & 15) * 8;
                async_copy16(&Ws[w][(size_t)(kt * 32 + r) * N + bn + c], &lW[w][r][c]);
            }
        async_wait<0>();
        __syncthreads();

        v16bf ax[2], as2[2];
        #pragma unroll
        for (int m = 0; m < 2; ++m) {
            const bf16* px = &lX[wm * 32 + m * 16 + l15][lh * 8];
            ((v8bf*)&ax[m])[0]  = *(const v8bf*)px;
            ((v8bf*)&ax[m])[1]  = *(const v8bf*)(px + 16);
            const bf16* ps = &lS[wm * 32 + m * 16 + l15][lh * 8];
            ((v8bf*)&as2[m])[0] = *(const v8bf*)ps;
            ((v8bf*)&as2[m])[1] = *(const v8bf*)(ps + 16);
        }
        #pragma unroll
        for (int n = 0; n < 4; ++n) {
            v16bf bw[4];
            #pragma unroll
            for (int w = 0; w < 4; ++w) {
                const bf16* p = &lW[w][lh * 8][wn * 64 + n * 16 + l15];
                #pragma unroll
                for (int s = 0; s < 16; ++s) bw[w][s] = p[(s < 8 ? s : s + 8) * 136];
            }
            #pragma unroll
            for (int m = 0; m < 2; ++m) {
                accz[m][n] = __builtin_amdgcn_wmma_f32_16x16x32_bf16(
                    false, ax[m],  false, bw[0], (short)0, accz[m][n], false, false);
                accz[m][n] = __builtin_amdgcn_wmma_f32_16x16x32_bf16(
                    false, as2[m], false, bw[1], (short)0, accz[m][n], false, false);
                acch[m][n] = __builtin_amdgcn_wmma_f32_16x16x32_bf16(
                    false, ax[m],  false, bw[2], (short)0, acch[m][n], false, false);
                acch[m][n] = __builtin_amdgcn_wmma_f32_16x16x32_bf16(
                    false, as2[m], false, bw[3], (short)0, acch[m][n], false, false);
            }
        }
        __syncthreads();
    }

    #pragma unroll
    for (int m = 0; m < 2; ++m)
        #pragma unroll
        for (int n = 0; n < 4; ++n)
            #pragma unroll
            for (int r = 0; r < 8; ++r) {
                int row = bm + wm * 32 + m * 16 + lh * 8 + r;
                int col = bn + wn * 64 + n * 16 + l15;
                float zp = accz[m][n][r] + bz[col];
                float hp = acch[m][n][r] + bh[col];
                float z  = 1.f / (1.f + expf(-zp));
                float ht = tanhf(hp);
                float s  = state[(size_t)row * UNITS + col];
                out[(size_t)row * UNITS + col] = (1.f - z) * s + z * ht;
            }
}

// ---------------------------------------------------------------------------
// launcher
// ---------------------------------------------------------------------------
extern "C" void kernel_launch(void* const* d_in, const int* in_sizes, int n_in,
                              void* d_out, int out_size, void* d_ws, size_t ws_size,
                              hipStream_t stream)
{
    const float* inputs = (const float*)d_in[0];   // [512,4096]
    const float* state  = (const float*)d_in[1];   // [512,256]
    const float* adj    = (const float*)d_in[2];   // [3,4096,4096]
    const float* wa     = (const float*)d_in[3];   // [3]
    const float* w_gcn  = (const float*)d_in[4];   // [4096,256]
    const float* wz     = (const float*)d_in[5];
    const float* uz     = (const float*)d_in[6];
    const float* bz     = (const float*)d_in[7];
    const float* wh     = (const float*)d_in[8];
    const float* uh     = (const float*)d_in[9];
    const float* bh     = (const float*)d_in[10];
    float* out = (float*)d_out;

    char* p = (char*)d_ws;
    bf16* adjmix = (bf16*)p; p += (size_t)NNODES * NNODES * 2;   // 32 MB (L2-resident)
    bf16* inb    = (bf16*)p; p += (size_t)BATCH * NNODES * 2;
    bf16* tb     = (bf16*)p; p += (size_t)BATCH * NNODES * 2;
    bf16* wgcnb  = (bf16*)p; p += (size_t)NNODES * UNITS * 2;
    bf16* xb     = (bf16*)p; p += (size_t)BATCH * UNITS * 2;
    bf16* stb    = (bf16*)p; p += (size_t)BATCH * UNITS * 2;
    bf16* wzb    = (bf16*)p; p += (size_t)UNITS * UNITS * 2;
    bf16* uzb    = (bf16*)p; p += (size_t)UNITS * UNITS * 2;
    bf16* whb    = (bf16*)p; p += (size_t)UNITS * UNITS * 2;
    bf16* uhb    = (bf16*)p; p += (size_t)UNITS * UNITS * 2;

    // adjacency mix + downconvert (bandwidth-bound floor of the whole op)
    k_adj_mix<<<(NNODES / 32) * (NNODES / 32), 256, 0, stream>>>(adj, wa, adjmix);

    // bf16 staging of GEMM operands
    k_cvt_bf16<<<(BATCH * NNODES) / 1024, 256, 0, stream>>>(inputs, inb);
    k_cvt_bf16<<<(NNODES * UNITS) / 1024, 256, 0, stream>>>(w_gcn, wgcnb);
    k_cvt_bf16<<<(BATCH * UNITS) / 1024, 256, 0, stream>>>(state, stb);
    k_cvt_bf16<<<(UNITS * UNITS) / 1024, 256, 0, stream>>>(wz, wzb);
    k_cvt_bf16<<<(UNITS * UNITS) / 1024, 256, 0, stream>>>(uz, uzb);
    k_cvt_bf16<<<(UNITS * UNITS) / 1024, 256, 0, stream>>>(wh, whb);
    k_cvt_bf16<<<(UNITS * UNITS) / 1024, 256, 0, stream>>>(uh, uhb);

    // T = inputs @ adj_mix            [512,4096] x [4096,4096]
    gemm_bf16<false><<<dim3(NNODES / 128, BATCH / 128), 256, 0, stream>>>(
        inb, adjmix, tb, BATCH, NNODES, NNODES);

    // x = relu(T @ w_gcn)             [512,4096] x [4096,256]
    gemm_bf16<true><<<dim3(UNITS / 128, BATCH / 128), 256, 0, stream>>>(
        tb, wgcnb, xb, BATCH, UNITS, NNODES);

    // fused GRU gates + output        [512,256]
    gru_kernel<<<dim3(UNITS / 128, BATCH / 128), 256, 0, stream>>>(
        xb, stb, wzb, uzb, whb, uhb, bz, bh, state, out);
}
